// SpatialAttentionModule_6313601925824
// MI455X (gfx1250) — matrix operand order repaired and checked by reading, not compiled, not run on hardware
//
#include <hip/hip_runtime.h>
#include <hip/hip_bf16.h>

// ---------------------------------------------------------------------------
// MI455X (gfx1250) fused spatial attention:
//   q/k/v projection (WMMA bf16) -> flash-attention (WMMA bf16, online
//   softmax, never materializes the 2x4096x4096 energy tensor) -> residual.
// K tiles streamed by the Tensor Data Mover (TENSORcnt), V tiles by async
// global->LDS copies (ASYNCcnt), compute on v_wmma_f32_16x16x32_bf16.
// Shapes: B=2, C=128, N=D*H*W=4096.
// ---------------------------------------------------------------------------

typedef __attribute__((ext_vector_type(16))) __bf16        v16bf;
typedef __attribute__((ext_vector_type(8)))  __bf16        v8bf;
typedef __attribute__((ext_vector_type(8)))  float         v8f;
typedef __attribute__((ext_vector_type(4)))  unsigned int  v4u;
typedef __attribute__((ext_vector_type(4)))  int           v4i;
typedef __attribute__((ext_vector_type(8)))  int           v8i;

// Address-space-qualified 16B vector pointers for the async-DMA builtins:
// clang's prototype is (v4i addrspace(1)*, v4i addrspace(3)*, imm, imm).
typedef __attribute__((address_space(1))) v4i g_v4i;   // global
typedef __attribute__((address_space(3))) v4i l_v4i;   // LDS

#define B_  2
#define C_  128
#define N_  4096
#define KT  32     // keys per tile (= WMMA K for the PV gemm)

#if __has_builtin(__builtin_amdgcn_global_load_async_to_lds_b128)
#define HAVE_ASYNC 1
#endif
#if __has_builtin(__builtin_amdgcn_tensor_load_to_lds)
#define HAVE_TDM 1
#endif

#if defined(HAVE_ASYNC)
// Per-lane async global->LDS 16B copy (tracked by ASYNCcnt).
static __device__ __forceinline__ void async_cp_b128(__bf16* ldst, const __bf16* gsrc) {
    __builtin_amdgcn_global_load_async_to_lds_b128(
        (g_v4i*)(unsigned long long)gsrc,
        (l_v4i*)(unsigned int)(unsigned long long)ldst,
        /*imm offset=*/0, /*cpol=*/0);
}
static __device__ __forceinline__ void async_wait0() {
#if __has_builtin(__builtin_amdgcn_s_wait_asynccnt)
    __builtin_amdgcn_s_wait_asynccnt(0);
#else
    asm volatile("s_wait_asynccnt 0x0" ::: "memory");
#endif
}
#endif

#if defined(HAVE_TDM)
// TDM 2D tile load: rows x row_elems bf16 elements, row pitch row_stride
// elements, contiguous into LDS.  Descriptor packed per CDNA5 ISA §8.3/8.4.
static __device__ __forceinline__ void tdm_load_2d(__bf16* lds_dst, const __bf16* gsrc,
                                                   int row_elems, int rows,
                                                   int tensor_d0, int tensor_d1,
                                                   int row_stride) {
    unsigned long long ga = (unsigned long long)gsrc;
    unsigned int       la = (unsigned int)(unsigned long long)lds_dst;
    v4u g0 = { 1u,                                    // count=1, user descriptor
               la,                                    // lds_addr [63:32]
               (unsigned int)(ga & 0xffffffffu),      // global_addr lo
               (unsigned int)((ga >> 32) & 0x01ffffffu) | (2u << 30) };  // hi | type=2
    v8i g1 = { (int)(1u << 16),                                        // data_size=2B
               (int)((tensor_d0 & 0xffff) << 16),                      // tensor_dim0 lo16
               (int)(((tensor_d0 >> 16) & 0xffff) |
                     ((tensor_d1 & 0xffff) << 16)),                    // d0 hi | d1 lo
               (int)(((tensor_d1 >> 16) & 0xffff) |
                     ((row_elems & 0xffff) << 16)),                    // d1 hi | tile_dim0
               (int)(rows & 0xffff),                                   // tile_dim1 (tile_dim2=0)
               (int)row_stride,                                        // dim0_stride lo32
               0, 0 };                                                 // stride hi / dim1_stride
    v4i gz = { 0, 0, 0, 0 };
#if __clang_major__ >= 23
    v8i gz8 = { 0, 0, 0, 0, 0, 0, 0, 0 };
    __builtin_amdgcn_tensor_load_to_lds(g0, g1, gz, gz, gz8, 0);
#else
    __builtin_amdgcn_tensor_load_to_lds(g0, g1, gz, gz, 0);
#endif
}
#endif

// Load two 16-byte bf16 chunks and concatenate into a 32-byte WMMA operand.
static __device__ __forceinline__ v16bf ld2x16B(const __bf16* p0, const __bf16* p1) {
    v8bf a = *(const v8bf*)p0;
    v8bf b = *(const v8bf*)p1;
    return __builtin_shufflevector(a, b, 0,1,2,3,4,5,6,7,8,9,10,11,12,13,14,15);
}

static __device__ __forceinline__ v8f wmma_bf16(v16bf a, v16bf b, v8f c) {
    // D(16x16 f32) = A(16x32 bf16) * B(32x16 bf16) + C
    return __builtin_amdgcn_wmma_f32_16x16x32_bf16(
        /*neg_a=*/false, a, /*neg_b=*/false, b,
        /*c_mod=*/(short)0, c, /*reuse_a=*/false, /*reuse_b=*/false);
}

// Reductions across the 16 lanes of a half-wave (the D-matrix stores one
// 16-wide row of scores per VGPR across lanes 0-15 / 16-31).
static __device__ __forceinline__ float half16_max(float v) {
    #pragma unroll
    for (int m = 8; m >= 1; m >>= 1) v = fmaxf(v, __shfl_xor(v, m, 16));
    return v;
}
static __device__ __forceinline__ float half16_sum(float v) {
    #pragma unroll
    for (int m = 8; m >= 1; m >>= 1) v += __shfl_xor(v, m, 16);
    return v;
}

// ---------------------------------------------------------------------------
// Prep: fp32 weights -> bf16 (row-major, which *is* the WMMA B-operand layout)
// ---------------------------------------------------------------------------
__global__ void cvt_weights_kernel(const float* __restrict__ Wq,
                                   const float* __restrict__ Wk,
                                   const float* __restrict__ Wv,
                                   __bf16* __restrict__ Wb) {
    int i = blockIdx.x * 256 + threadIdx.x;           // 0 .. 16383
    if (i < C_ * C_) {
        Wb[i]             = (__bf16)Wq[i];
        Wb[C_*C_ + i]     = (__bf16)Wk[i];
        Wb[2*C_*C_ + i]   = (__bf16)Wv[i];
    }
}

// Prep: x (B,C,N) fp32 -> xT (B,N,C) bf16 for the projection A operand.
__global__ void xpose_kernel(const float* __restrict__ x,
                             __bf16* __restrict__ xT) {
    int i = blockIdx.x * 256 + threadIdx.x;           // over B*N*C
    int c = i % C_;
    int n = (i / C_) % N_;
    int b = i / (C_ * N_);
    xT[i] = (__bf16)x[((size_t)b * C_ + c) * N_ + n];
}

// ---------------------------------------------------------------------------
// QKV projection: per (batch, 16-query tile), wave w computes output-channel
// tile [16w,16w+16) for q, k and v.  12 WMMAs per wave.
//   qT,kT stored (B,N,C) bf16   (A operand layout for the S gemm / B for K)
//   v     stored (B,C,N) bf16   (B operand layout for the PV gemm)
// ---------------------------------------------------------------------------
__global__ __launch_bounds__(256)
void qkv_proj_kernel(const __bf16* __restrict__ xT,
                     const __bf16* __restrict__ Wb,
                     const float* __restrict__ bq,
                     const float* __restrict__ bk,
                     const float* __restrict__ bv,
                     __bf16* __restrict__ qT,
                     __bf16* __restrict__ kT,
                     __bf16* __restrict__ vB) {
    const int b    = blockIdx.y;
    const int n0   = blockIdx.x * 16;
    const int wave = threadIdx.x >> 5;
    const int lane = threadIdx.x & 31;
    const int lm   = lane & 15;
    const int hi   = lane >> 4;
    const int o0   = wave * 16;

    // A operand: xT rows. Lane = query row; K chunks split across lane halves
    // contiguously ({0-15} lo lanes / {16-31} hi), matching the B operand.
    const __bf16* xrow = xT + ((size_t)b * N_ + n0 + lm) * C_;
    v16bf A[4];
    #pragma unroll
    for (int kc = 0; kc < 4; ++kc)
        A[kc] = ld2x16B(xrow + kc*32 + 16*hi, xrow + kc*32 + 16*hi + 8);

    const int oc = o0 + lm;   // this lane's output channel (B-operand column)
    #pragma unroll
    for (int m = 0; m < 3; ++m) {
        const __bf16* wrow = Wb + m * C_ * C_ + (size_t)oc * C_;
        v8f acc = {};
        #pragma unroll
        for (int kc = 0; kc < 4; ++kc) {
            v16bf Bv = *(const v16bf*)(wrow + kc*32 + 16*hi); // 32B aligned
            acc = wmma_bf16(A[kc], Bv, acc);
        }
        const float bias = (m == 0 ? bq : (m == 1 ? bk : bv))[oc];
        if (m == 2) {
            // v in (C,N): lane's column is channel, D rows are consecutive n.
            __bf16* vout = vB + ((size_t)b * C_ + oc) * N_ + n0 + 8*hi;
            #pragma unroll
            for (int r = 0; r < 8; ++r) vout[r] = (__bf16)(acc[r] + bias);
        } else {
            __bf16* outp = (m == 0 ? qT : kT);
            #pragma unroll
            for (int r = 0; r < 8; ++r)
                outp[((size_t)b * N_ + n0 + r + 8*hi) * C_ + oc] =
                    (__bf16)(acc[r] + bias);
        }
    }
}

// ---------------------------------------------------------------------------
// Flash attention: block = 8 waves = 128 queries; each wave owns 16 query
// rows.  K tile DMA'd by the TDM, V tile async-copied to LDS, online softmax
// in registers, O accumulated in f32.
// ---------------------------------------------------------------------------
__global__ __launch_bounds__(256)
void attn_kernel(const __bf16* __restrict__ qT,
                 const __bf16* __restrict__ kT,
                 const __bf16* __restrict__ vB,
                 const float* __restrict__ xin,
                 const float* __restrict__ gamma,
                 float* __restrict__ out) {
    __shared__ __align__(32) __bf16 ktile[KT][C_];     // [key][channel]   8KB
    __shared__ __align__(32) __bf16 vtile[C_][KT];     // [channel][key]   8KB
    __shared__ __align__(32) __bf16 ptile[8][16][KT];  // per-wave P xpose 8KB

    const int b    = blockIdx.y;
    const int wave = threadIdx.x >> 5;
    const int lane = threadIdx.x & 31;
    const int lm   = lane & 15;
    const int hi   = lane >> 4;
    const int n0   = blockIdx.x * 128 + wave * 16;     // this wave's queries

    // Q tile (A operand) lives in registers for the whole kernel.
    const __bf16* qrow = qT + ((size_t)b * N_ + n0 + lm) * C_;
    v16bf Qa[4];
    #pragma unroll
    for (int kc = 0; kc < 4; ++kc)
        Qa[kc] = ld2x16B(qrow + kc*32 + 16*hi, qrow + kc*32 + 16*hi + 8);

    v8f O[8];                       // 16 queries x 128 channels, f32
    #pragma unroll
    for (int t = 0; t < 8; ++t) O[t] = {};
    float mrun[8], lrun[8];
    #pragma unroll
    for (int r = 0; r < 8; ++r) { mrun[r] = -3.0e38f; lrun[r] = 0.f; }

    // Staging work split: K tile = 32 rows x 256B, V tile = 128 rows x 64B.
    const int tid = threadIdx.x;
    const int krow = tid >> 3, kseg = tid & 7;   (void)krow; (void)kseg;
    const int vrow = tid >> 1, vseg = tid & 1;
    const __bf16* kbase = kT + (size_t)b * N_ * C_;
    const __bf16* vbase = vB + (size_t)b * C_ * N_;

    #pragma unroll 1
    for (int m0 = 0; m0 < N_; m0 += KT) {
        __syncthreads();   // everyone done reading previous K/V tiles

        // ---- K tile [32 keys][128 c] via Tensor Data Mover (wave 0 only)
#if defined(HAVE_TDM)
        if (wave == 0)
            tdm_load_2d(&ktile[0][0], kbase + (size_t)m0 * C_,
                        /*row_elems=*/C_, /*rows=*/KT,
                        /*tensor_d0=*/C_, /*tensor_d1=*/N_,
                        /*row_stride=*/C_);
#else
        *(v16bf*)&ktile[krow][kseg * 16] =
            *(const v16bf*)(kbase + (size_t)(m0 + krow) * C_ + kseg * 16);
#endif

        // ---- V tile [128 c][32 keys] via async global->LDS (all threads)
        {
            const __bf16* vsrc = vbase + (size_t)vrow * N_ + m0 + vseg * 16;
            __bf16*       vdst = &vtile[vrow][vseg * 16];
#if defined(HAVE_ASYNC)
            async_cp_b128(vdst,     vsrc);
            async_cp_b128(vdst + 8, vsrc + 8);
#else
            *(v16bf*)vdst = *(const v16bf*)vsrc;
#endif
        }
        if (m0 + KT < N_) {   // warm L2 for the next tile
            __builtin_prefetch(kbase + (size_t)(m0 + KT) * C_ + tid * 16, 0, 1);
            __builtin_prefetch(vbase + (size_t)vrow * N_ + m0 + KT + vseg * 16, 0, 1);
        }

#if defined(HAVE_TDM)
        if (wave == 0) __builtin_amdgcn_s_wait_tensorcnt(0);
#endif
#if defined(HAVE_ASYNC)
        async_wait0();
#endif
        __syncthreads();

        // ---- S(16q x 32k) = Qa * Ktile : 8 WMMAs (4 c-chunks x 2 key halves)
        v8f S0 = {}, S1 = {};
        #pragma unroll
        for (int kc = 0; kc < 4; ++kc) {
            v16bf B0 = *(const v16bf*)&ktile[lm][kc*32 + 16*hi];
            v16bf B1 = *(const v16bf*)&ktile[16 + lm][kc*32 + 16*hi];
            S0 = wmma_bf16(Qa[kc], B0, S0);
            S1 = wmma_bf16(Qa[kc], B1, S1);
        }

        // ---- online softmax; row (r + 8*hi) spans the 16 lanes of a half.
        float P0[8], P1[8];
        #pragma unroll
        for (int r = 0; r < 8; ++r) {
            float mloc  = half16_max(fmaxf(S0[r], S1[r]));
            float mnew  = fmaxf(mrun[r], mloc);
            float scale = __expf(mrun[r] - mnew);
            mrun[r] = mnew;
            P0[r] = __expf(S0[r] - mnew);
            P1[r] = __expf(S1[r] - mnew);
            lrun[r] = lrun[r] * scale + half16_sum(P0[r] + P1[r]);
            #pragma unroll
            for (int t = 0; t < 8; ++t) O[t][r] = O[t][r] * scale;
        }

        // ---- transpose P (D layout -> A layout) through wave-private LDS.
        // Same-wave DS ops retire in order, so no barrier is required.
        #pragma unroll
        for (int r = 0; r < 8; ++r) {
            ptile[wave][r + 8*hi][lm]      = (__bf16)P0[r];
            ptile[wave][r + 8*hi][16 + lm] = (__bf16)P1[r];
        }
        const __bf16* prow = &ptile[wave][lm][0];
        v16bf Pa = ld2x16B(prow + 16*hi, prow + 16*hi + 8);

        // ---- O(16q x 128c) += P(16x32) * Vtile(32 x 128) : 8 WMMAs
        #pragma unroll
        for (int t = 0; t < 8; ++t) {
            v16bf Vv = *(const v16bf*)&vtile[16*t + lm][16*hi];
            O[t] = wmma_bf16(Pa, Vv, O[t]);
        }
    }

    // ---- epilogue: out = x + gamma * (O / l)
    const float g = gamma[0];
    #pragma unroll
    for (int t = 0; t < 8; ++t) {
        const int c = 16*t + lm;
        const size_t base = ((size_t)b * C_ + c) * N_ + n0 + 8*hi;
        #pragma unroll
        for (int r = 0; r < 8; ++r)
            out[base + r] = xin[base + r] + g * (O[t][r] / lrun[r]);
    }
}

// ---------------------------------------------------------------------------
extern "C" void kernel_launch(void* const* d_in, const int* in_sizes, int n_in,
                              void* d_out, int out_size, void* d_ws, size_t ws_size,
                              hipStream_t stream) {
    (void)in_sizes; (void)n_in; (void)out_size; (void)ws_size;
    const float* x     = (const float*)d_in[0];
    const float* Wq    = (const float*)d_in[1];
    const float* bq    = (const float*)d_in[2];
    const float* Wk    = (const float*)d_in[3];
    const float* bk    = (const float*)d_in[4];
    const float* Wv    = (const float*)d_in[5];
    const float* bv    = (const float*)d_in[6];
    const float* gamma = (const float*)d_in[7];
    float* out = (float*)d_out;

    // Workspace carve-up (bf16): weights 96KB, xT/qT/kT/v 2MB each -> ~8.1MB.
    __bf16* wsb = (__bf16*)d_ws;
    __bf16* Wb  = wsb;                               // 3 * 128*128
    __bf16* xT  = Wb + 3 * C_ * C_;                  // (B,N,C)
    __bf16* qT  = xT + (size_t)B_ * N_ * C_;         // (B,N,C)
    __bf16* kT  = qT + (size_t)B_ * N_ * C_;         // (B,N,C)
    __bf16* vB  = kT + (size_t)B_ * N_ * C_;         // (B,C,N)

    cvt_weights_kernel<<<(C_*C_ + 255)/256, 256, 0, stream>>>(Wq, Wk, Wv, Wb);
    xpose_kernel<<<(B_*N_*C_)/256, 256, 0, stream>>>(x, xT);
    qkv_proj_kernel<<<dim3(N_/16, B_), 256, 0, stream>>>(xT, Wb, bq, bk, bv, qT, kT, vB);
    attn_kernel<<<dim3(N_/128, B_), 256, 0, stream>>>(qT, kT, vB, x, gamma, out);
}
